// Zeus_25443386262249
// MI455X (gfx1250) — matrix operand order repaired and checked by reading, not compiled, not run on hardware
//
#include <hip/hip_runtime.h>

typedef __attribute__((ext_vector_type(16))) _Float16 v16h;
typedef __attribute__((ext_vector_type(8)))  _Float16 v8h;
typedef __attribute__((ext_vector_type(4)))  _Float16 v4h;
typedef __attribute__((ext_vector_type(8)))  float    v8f;
typedef __attribute__((ext_vector_type(4)))  float    v4f;

#define D_FEAT      512
#define N_OUT       1024
#define SEQ         8192
#define ROWS_PER_WG 32
#define A_STRIDE_H  520    // halves per LDS row (pad: 4-bank skew, conflict-free)
#define KV_STRIDE_F 1028   // floats per LDS KV row (pad: 4-bank skew)

union AF { v16h f; v8h h[2]; };

__device__ __forceinline__ float wave_sum(float x) {
#pragma unroll
  for (int off = 16; off >= 1; off >>= 1) x += __shfl_xor(x, off, 32);
  return x;
}

// ---------------- prep: W fp32 -> f16 (row-major 1024 x 512) ----------------
__global__ void prep_kernel(const float* __restrict__ W, _Float16* __restrict__ Wh) {
  int i = blockIdx.x * 256 + threadIdx.x;           // 131072 float4 groups
  v4f w = *(const v4f*)(W + (size_t)i * 4);
  v4h h;
#pragma unroll
  for (int j = 0; j < 4; ++j) h[j] = (_Float16)w[j];
  *(v4h*)(Wh + (size_t)i * 4) = h;
}

// ------------- fused LN + GEMM(WMMA f16) + RoPE + norm-reduce ---------------
__global__ void __launch_bounds__(256, 1)
fused_kernel(const float* __restrict__ X, const _Float16* __restrict__ Wh,
             const float* __restrict__ b_in, const float* __restrict__ gamma,
             const float* __restrict__ beta, float* __restrict__ partials) {
  extern __shared__ char smem[];
  _Float16* a_lds  = (_Float16*)smem;                                   // 32 x 520 halves (phase A/B)
  float*    kv_lds = (float*)smem;                                      // 32 x 1028 f32 (phase C/D, union)
  float*    a_part = (float*)(smem + ROWS_PER_WG * KV_STRIDE_F * 4);    // 8 x 512 f32 per-wave partials

  const int tid   = threadIdx.x;
  const int wave  = tid >> 5;
  const int lane  = tid & 31;
  const int lhalf = lane >> 4;   // 0|1
  const int lmod  = lane & 15;
  const int wg    = blockIdx.x;
  const int row_base = wg * ROWS_PER_WG;
  const int n_base   = wave * 128;

  // ---- Phase A: LayerNorm, 4 rows per wave, write f16 tile to LDS ----
#pragma unroll
  for (int rr = 0; rr < 4; ++rr) {
    int r = wave * 4 + rr;
    const float* xr = X + (size_t)(row_base + r) * D_FEAT;
    v4f xv[4];
    float s = 0.f, s2 = 0.f;
#pragma unroll
    for (int c = 0; c < 4; ++c) {
      xv[c] = *(const v4f*)(xr + c * 128 + lane * 4);
#pragma unroll
      for (int j = 0; j < 4; ++j) { float t = xv[c][j]; s += t; s2 += t * t; }
    }
    s = wave_sum(s); s2 = wave_sum(s2);
    float mu = s * (1.f / 512.f);
    float rs = rsqrtf(s2 * (1.f / 512.f) - mu * mu + 1e-5f);
#pragma unroll
    for (int c = 0; c < 4; ++c) {
      int d0 = c * 128 + lane * 4;
      v4f g  = *(const v4f*)(gamma + d0);
      v4f bt = *(const v4f*)(beta + d0);
      v4h hv;
#pragma unroll
      for (int j = 0; j < 4; ++j) hv[j] = (_Float16)((xv[c][j] - mu) * rs * g[j] + bt[j]);
      *(v4h*)(a_lds + r * A_STRIDE_H + d0) = hv;
    }
  }
  __syncthreads();

  // ---- Phase B: GEMM. Wave owns n in [wave*128, wave*128+128), all 32 rows ----
  v8f acc[8][2];
#pragma unroll
  for (int j = 0; j < 8; ++j)
#pragma unroll
    for (int mt = 0; mt < 2; ++mt)
#pragma unroll
      for (int q = 0; q < 8; ++q) acc[j][mt][q] = 0.f;

#pragma unroll 2
  for (int k0 = 0; k0 < 512; k0 += 32) {
    AF af[2];
#pragma unroll
    for (int mt = 0; mt < 2; ++mt) {
      // ISA 16-bit A layout: lane L -> row L&15; halves k0+8*(L>>4)..+7 and +16
      const _Float16* ap = a_lds + (mt * 16 + lmod) * A_STRIDE_H + k0 + lhalf * 8;
      af[mt].h[0] = *(const v8h*)ap;
      af[mt].h[1] = *(const v8h*)(ap + 16);
    }
#pragma unroll
    for (int j = 0; j < 8; ++j) {
      // ISA 16-bit B layout: lane L -> column n0+(L&15); halves k0+8*(L>>4)..+7 and +16
      const _Float16* bp = Wh + (size_t)(n_base + j * 16 + lmod) * 512 + k0 + lhalf * 8;
      AF bf;
      bf.h[0] = *(const v8h*)bp;
      bf.h[1] = *(const v8h*)(bp + 16);
#pragma unroll
      for (int mt = 0; mt < 2; ++mt)
        acc[j][mt] = __builtin_amdgcn_wmma_f32_16x16x32_f16(
            false, af[mt].f, false, bf.f, (short)0, acc[j][mt], false, false);
    }
  }
  __syncthreads();  // all a_lds reads done; kv_lds may now overwrite the union

  // ---- Phase C: bias + RoPE in registers, spill KV f32 tile to LDS ----
  const int sp = row_base & (SEQ - 1);  // seq position of row m is sp + m
#pragma unroll
  for (int j = 0; j < 8; ++j) {
    int n = n_base + j * 16 + lmod;     // column in [0,1024)
    float bias = b_in[n];
    int dd = n & 511;                   // dim inside K or V
    bool do_rope = dd < 256;            // uniform per 16-wide tile
    float sgn = (dd & 1) ? 1.f : -1.f;
    float inv_freq = do_rope ? __expf(-(float)(dd & ~1) * (9.210340372f / 256.0f)) : 0.f;
#pragma unroll
    for (int mt = 0; mt < 2; ++mt) {
#pragma unroll
      for (int v = 0; v < 8; ++v) {
        int m = mt * 16 + v + 8 * lhalf;        // ISA C layout: M = vgpr + 8*(lane>>4)
        float xp = acc[j][mt][v] + bias;
        float x = xp;
        if (do_rope) {
          float partner = __shfl_xor(xp, 1, 32);  // adjacent N = adjacent lane
          float ang = (float)(sp + m) * inv_freq;
          float c, sn;
          __sincosf(ang, &sn, &c);
          x = xp * c + sgn * partner * sn;        // even: x*c - x[d+1]*s; odd: x*c + x[d-1]*s
        }
        kv_lds[m * KV_STRIDE_F + n] = x;
      }
    }
  }
  __syncthreads();

  // ---- Phase D: per-row l2 norms + K.V accumulation (4 rows per wave) ----
  float a_acc[16];
#pragma unroll
  for (int q = 0; q < 16; ++q) a_acc[q] = 0.f;
#pragma unroll
  for (int rr = 0; rr < 4; ++rr) {
    int r = wave * 4 + rr;
    float p[16];
    float sk = 0.f, sv = 0.f;
#pragma unroll
    for (int c = 0; c < 4; ++c) {
      int d0 = c * 128 + lane * 4;
      v4f kx = *(const v4f*)(kv_lds + r * KV_STRIDE_F + d0);
      v4f vx = *(const v4f*)(kv_lds + r * KV_STRIDE_F + 512 + d0);
#pragma unroll
      for (int jj = 0; jj < 4; ++jj) {
        sk += kx[jj] * kx[jj];
        sv += vx[jj] * vx[jj];
        p[c * 4 + jj] = kx[jj] * vx[jj];
      }
    }
    sk = wave_sum(sk); sv = wave_sum(sv);
    float rn = rsqrtf(sk * sv);        // 1/(|K||V|)
#pragma unroll
    for (int q = 0; q < 16; ++q) a_acc[q] += p[q] * rn;
  }
  // per-wave slice (no atomics -> deterministic)
#pragma unroll
  for (int c = 0; c < 4; ++c) {
    int d0 = c * 128 + lane * 4;
    v4f t;
#pragma unroll
    for (int jj = 0; jj < 4; ++jj) t[jj] = a_acc[c * 4 + jj];
    *(v4f*)(a_part + wave * 512 + d0) = t;
  }
  __syncthreads();

  // ---- Phase E: fixed-order cross-wave sum -> per-WG partial to global ----
  if (tid < 128) {
    int d0 = tid * 4;
    v4f s = *(const v4f*)(a_part + d0);
#pragma unroll
    for (int w = 1; w < 8; ++w) {
      v4f t = *(const v4f*)(a_part + w * 512 + d0);
#pragma unroll
      for (int jj = 0; jj < 4; ++jj) s[jj] += t[jj];
    }
    *(v4f*)(partials + (size_t)wg * 512 + d0) = s;
  }
}

// ---------------- reduce: sum 256 WG partials per batch, fixed order ----------------
__global__ void reduce_kernel(const float* __restrict__ partials, float* __restrict__ Afin) {
  int idx = blockIdx.x * 256 + threadIdx.x;  // 0..4095
  int b = idx >> 9, d = idx & 511;
  const float* p = partials + (size_t)b * 256 * 512 + d;
  float s = 0.f;
#pragma unroll 8
  for (int w = 0; w < 256; ++w) s += p[(size_t)w * 512];
  Afin[idx] = 0.5f * s + 0.5f;
}

// ---------------- scale: out = A[b,d] * X ----------------
__global__ void scale_kernel(const float* __restrict__ X, const float* __restrict__ Afin,
                             float* __restrict__ out) {
  size_t i = (size_t)blockIdx.x * 256 + threadIdx.x;
  size_t base = i * 4;
  int d = (int)(base & 511);
  int b = (int)(base >> 22);                 // 8192*512 = 2^22 elems per batch
  v4f x = *(const v4f*)(X + base);
  v4f a = *(const v4f*)(Afin + b * 512 + d);
  v4f r;
#pragma unroll
  for (int j = 0; j < 4; ++j) r[j] = x[j] * a[j];
  *(v4f*)(out + base) = r;
}

extern "C" void kernel_launch(void* const* d_in, const int* in_sizes, int n_in,
                              void* d_out, int out_size, void* d_ws, size_t ws_size,
                              hipStream_t stream) {
  const float* X     = (const float*)d_in[0];
  const float* W_in  = (const float*)d_in[1];
  const float* b_in  = (const float*)d_in[2];
  const float* gamma = (const float*)d_in[3];
  const float* beta  = (const float*)d_in[4];
  float* out = (float*)d_out;

  char* ws = (char*)d_ws;
  _Float16* Wh     = (_Float16*)ws;                                   // 1 MB
  float* partials  = (float*)(ws + (1 << 20));                        // 2048*512*4 = 4 MB
  float* Afin      = (float*)(ws + (1 << 20) + 2048 * 512 * 4);       // 16 KB

  prep_kernel<<<512, 256, 0, stream>>>(W_in, Wh);

  size_t shmem = (size_t)ROWS_PER_WG * KV_STRIDE_F * sizeof(float)    // 131584 B union
               + 8 * 512 * sizeof(float);                             // 16384 B wave partials
  fused_kernel<<<2048, 256, shmem, stream>>>(X, Wh, b_in, gamma, beta, partials);

  reduce_kernel<<<16, 256, 0, stream>>>(partials, Afin);

  scale_kernel<<<32768, 256, 0, stream>>>(X, Afin, out);
}
